// scaled_dot_product_attention_36902359007751
// MI455X (gfx1250) — compile-verified, matrix operand order checked by reading, not run on hardware
//
#include <hip/hip_runtime.h>

// ---------------------------------------------------------------------------
// Problem constants (match reference: B=4, S=2048, D=1024)
// ---------------------------------------------------------------------------
constexpr int BATCH = 4;
constexpr int SEQ   = 2048;
constexpr int DIM   = 1024;
constexpr int MTOT  = BATCH * SEQ;          // 8192 rows total
constexpr float NEGV  = -1.0e10f;
constexpr float SCALE = 0.03125f;           // 1/sqrt(1024)

// ---------------------------------------------------------------------------
// Vector / WMMA types (trivial ext-vectors so unions stay trivial)
// ---------------------------------------------------------------------------
typedef __attribute__((ext_vector_type(16))) __bf16        v16bf;
typedef __attribute__((ext_vector_type(8)))  float         v8f;
typedef __attribute__((ext_vector_type(4)))  float         v4f;
typedef __attribute__((ext_vector_type(4)))  unsigned int  v4u;
typedef __attribute__((ext_vector_type(2)))  unsigned int  v2u;
typedef __attribute__((ext_vector_type(8)))  int           v8i;
typedef __attribute__((ext_vector_type(4)))  int           v4i;

union FragBF {               // 32 bytes: one WMMA bf16 operand per lane
    v16bf v;
    v4u   q[2];
};

__device__ __forceinline__ unsigned short f2bf(float f) {
    union { float f; unsigned int u; } x; x.f = f;
    unsigned int r = x.u + 0x7FFFu + ((x.u >> 16) & 1u);   // round-nearest-even
    return (unsigned short)(r >> 16);
}

__device__ __forceinline__ v16bf ld_frag(const unsigned short* p) {
    FragBF f;
    const v4u* q = reinterpret_cast<const v4u*>(p);
    f.q[0] = q[0];
    f.q[1] = q[1];
    return f.v;
}

__device__ __forceinline__ v8f wmma_bf16(v16bf a, v16bf b, v8f c) {
    return __builtin_amdgcn_wmma_f32_16x16x32_bf16(false, a, false, b,
                                                   (short)0, c, false, false);
}

__device__ __forceinline__ unsigned rfl(unsigned x) {
    return (unsigned)__builtin_amdgcn_readfirstlane((int)x);
}

// ---------------------------------------------------------------------------
// TDM: issue a 2D bf16 tile load (tile_dim0=32 elems, tile_dim1=128 rows)
// global -> LDS with 16B pad after each 64B row (LDS row stride = 80B,
// matching the [128][40]-short fragment layout).  D# per ISA 08_async_tensor.
// Must be issued by a full wave (EXEC ignored, uses TENSORcnt).
// Toolchain: 6-arg builtin (clang-23 / therock-10.0).
// ---------------------------------------------------------------------------
__device__ __forceinline__ void tdm_load_tile32x128(unsigned lds_off,
                                                    const unsigned short* gptr,
                                                    unsigned tensor_d1) {
    unsigned long long ga = (unsigned long long)(uintptr_t)gptr;   // byte addr
    v4u g0;
    g0[0] = rfl(1u);                                  // count=1, user mode
    g0[1] = rfl(lds_off);                             // LDS byte address
    g0[2] = rfl((unsigned)(ga & 0xFFFFFFFFu));        // global addr lo
    g0[3] = rfl((unsigned)((ga >> 32) & 0x1FFFFFFu) | (2u << 30)); // hi + type=2

    v8i g1;
    // data_size=2B(code1), pad_enable, pad_interval=16DW(code3), pad_amount=4DW(code3)
    g1[0] = (int)rfl((1u << 16) | (1u << 20) | (3u << 22) | (3u << 25));
    g1[1] = (int)rfl(((unsigned)DIM & 0xFFFFu) << 16);         // tensor_dim0 lo16
    g1[2] = (int)rfl(((unsigned)DIM >> 16) |
                     ((tensor_d1 & 0xFFFFu) << 16));           // dim0 hi | dim1 lo
    g1[3] = (int)rfl((tensor_d1 >> 16) | (32u << 16));         // dim1 hi | tile_dim0=32
    g1[4] = (int)rfl(128u);                                    // tile_dim1=128, tile_dim2=0
    g1[5] = (int)rfl((unsigned)DIM);                           // tensor_dim0_stride lo32
    g1[6] = 0;                                                 // stride hi / dim1_stride
    g1[7] = 0;
    v4i z4 = {0, 0, 0, 0};
    v8i z8 = {0, 0, 0, 0, 0, 0, 0, 0};
    __builtin_amdgcn_tensor_load_to_lds(g0, g1, z4, z4, z8, 0);
}

// ---------------------------------------------------------------------------
// Kernel 0: elementwise fp32 -> bf16 (feeds the TDM/WMMA bf16 pipeline)
// ---------------------------------------------------------------------------
__global__ __launch_bounds__(256)
void cvt_bf16_kernel(const float* __restrict__ src,
                     unsigned short* __restrict__ dst, int n4) {
    int i = blockIdx.x * 256 + threadIdx.x;
    if (i < n4) {
        v4f f = reinterpret_cast<const v4f*>(src)[i];
        v2u p = {(unsigned)f2bf(f[0]) | ((unsigned)f2bf(f[1]) << 16),
                 (unsigned)f2bf(f[2]) | ((unsigned)f2bf(f[3]) << 16)};
        reinterpret_cast<v2u*>(dst)[i] = p;
    }
}

// ---------------------------------------------------------------------------
// Kernel 1: projection GEMM  out[m,n](bf16) = sum_k A[m,k]*W[n,k] + bias[n]
//   A: [MTOT, DIM] bf16, W: [DIM, DIM] bf16 (n = row), bias fp32.
//   TDM double-buffered LDS tiles; 8 waves, wave tile 32x64; BK=32.
//   TRANSPOSE_OUT stores out[n*MTOT+m] (used for V so attention reads
//   contiguous key-dim fragments for the P*V WMMA B operand).
// ---------------------------------------------------------------------------
template <bool TRANSPOSE_OUT>
__global__ __launch_bounds__(256)
void qkv_proj_kernel(const unsigned short* __restrict__ A,
                     const unsigned short* __restrict__ W,
                     const float* __restrict__ bias,
                     unsigned short* __restrict__ out) {
    __shared__ __align__(16) unsigned short sA[2][128][40];   // 80B row stride
    __shared__ __align__(16) unsigned short sW[2][128][40];

    const int tid  = threadIdx.x;
    const int lane = tid & 31;
    const int wv   = tid >> 5;
    const int lo   = lane & 15;
    const int hi   = lane >> 4;
    const int m0   = blockIdx.x * 128;
    const int n0   = blockIdx.y * 128;
    const int wave_m = (wv & 3) * 32;
    const int wave_n = (wv >> 2) * 64;

    v8f acc[2][4];
    const v8f vzero = {0.f,0.f,0.f,0.f,0.f,0.f,0.f,0.f};
#pragma unroll
    for (int mt = 0; mt < 2; ++mt)
#pragma unroll
        for (int nt = 0; nt < 4; ++nt) acc[mt][nt] = vzero;

    float bias_v[4];
#pragma unroll
    for (int nt = 0; nt < 4; ++nt)
        bias_v[nt] = bias[n0 + wave_n + nt * 16 + lo];

    constexpr int NKB = DIM / 32;

    if (wv == 0) {   // prologue: DMA first K-tile of both operands
        tdm_load_tile32x128((unsigned)(uintptr_t)&sA[0][0][0],
                            A + (size_t)m0 * DIM, (unsigned)MTOT);
        tdm_load_tile32x128((unsigned)(uintptr_t)&sW[0][0][0],
                            W + (size_t)n0 * DIM, (unsigned)DIM);
    }

    for (int kb = 0; kb < NKB; ++kb) {
        const int buf = kb & 1;
        if (wv == 0) {
            if (kb + 1 < NKB) {   // DMA next tile into the other buffer
                tdm_load_tile32x128((unsigned)(uintptr_t)&sA[buf ^ 1][0][0],
                                    A + (size_t)m0 * DIM + (kb + 1) * 32,
                                    (unsigned)MTOT);
                tdm_load_tile32x128((unsigned)(uintptr_t)&sW[buf ^ 1][0][0],
                                    W + (size_t)n0 * DIM + (kb + 1) * 32,
                                    (unsigned)DIM);
                __builtin_amdgcn_s_wait_tensorcnt(2);  // current buffer complete
            } else {
                __builtin_amdgcn_s_wait_tensorcnt(0);
            }
        }
        __syncthreads();          // current buffer visible to all waves

        v16bf afr[2], bfr[4];
#pragma unroll
        for (int mt = 0; mt < 2; ++mt)
            afr[mt] = ld_frag(&sA[buf][wave_m + mt * 16 + lo][hi * 16]);
#pragma unroll
        for (int nt = 0; nt < 4; ++nt)
            bfr[nt] = ld_frag(&sW[buf][wave_n + nt * 16 + lo][hi * 16]);

#pragma unroll
        for (int mt = 0; mt < 2; ++mt)
#pragma unroll
            for (int nt = 0; nt < 4; ++nt)
                acc[mt][nt] = wmma_bf16(afr[mt], bfr[nt], acc[mt][nt]);

        __syncthreads();          // all reads done before TDM overwrites
    }

    // epilogue: bias + bf16 store (C layout: VGPR r -> M=r (+8 for hi lanes))
#pragma unroll
    for (int mt = 0; mt < 2; ++mt)
#pragma unroll
        for (int nt = 0; nt < 4; ++nt)
#pragma unroll
            for (int r = 0; r < 8; ++r) {
                int m_l = wave_m + mt * 16 + r + hi * 8;
                int n_l = wave_n + nt * 16 + lo;
                float vvl = acc[mt][nt][r] + bias_v[nt];
                size_t idx = TRANSPOSE_OUT
                                 ? (size_t)(n0 + n_l) * MTOT + (m0 + m_l)
                                 : (size_t)(m0 + m_l) * DIM + (n0 + n_l);
                out[idx] = f2bf(vvl);
            }
}

// ---------------------------------------------------------------------------
// Kernel 2: flash attention with online softmax.
//   q,k: bf16 [MTOT, DIM] row-major.  vT: bf16 [DIM, MTOT] (transposed).
//   Block: one batch b, 32 query rows; 8 waves; wave w owns output columns
//   [w*128, w*128+128).  QK^T reduction over DIM split across waves
//   (128-deep slice each), combined via ds_add_f32 atomics in LDS.
// ---------------------------------------------------------------------------
__global__ __launch_bounds__(256)
void flash_attn_kernel(const unsigned short* __restrict__ q,
                       const unsigned short* __restrict__ k,
                       const unsigned short* __restrict__ vT,
                       const int* __restrict__ mask,
                       float* __restrict__ out) {
    __shared__ __align__(16) float          s_scores[32][36];
    __shared__ __align__(16) unsigned short s_p[32][40];
    __shared__ float s_corr[32];
    __shared__ float s_m[32];
    __shared__ float s_l[32];

    const int tid  = threadIdx.x;
    const int lane = tid & 31;
    const int wv   = tid >> 5;
    const int lo   = lane & 15;
    const int hi   = lane >> 4;
    const int qt   = blockIdx.x;
    const int b    = blockIdx.y;
    const int q0   = qt * 32;
    const int dbase = wv * 128;

    if (tid < 32) { s_m[tid] = -3.0e38f; s_l[tid] = 0.0f; }

    v8f acc[2][8];
    const v8f vzero = {0.f,0.f,0.f,0.f,0.f,0.f,0.f,0.f};
#pragma unroll
    for (int mt = 0; mt < 2; ++mt)
#pragma unroll
        for (int nt = 0; nt < 8; ++nt) acc[mt][nt] = vzero;

    v16bf qfr[2][4];
#pragma unroll
    for (int mt = 0; mt < 2; ++mt)
#pragma unroll
        for (int ks = 0; ks < 4; ++ks) {
            size_t off = (size_t)(b * SEQ + q0 + mt * 16 + lo) * DIM +
                         dbase + ks * 32 + hi * 16;
            qfr[mt][ks] = ld_frag(q + off);
        }
    __syncthreads();

    float* sc_flat = &s_scores[0][0];

    for (int kt = 0; kt <= qt; ++kt) {   // causal: skip tiles above diagonal
        const int k0 = kt * 32;

        for (int i = tid; i < 32 * 36; i += 256) sc_flat[i] = 0.0f;
        __syncthreads();

        // ---- partial QK^T over this wave's 128-deep D slice ----
#pragma unroll
        for (int nt = 0; nt < 2; ++nt) {
            v16bf kfr[4];
#pragma unroll
            for (int ks = 0; ks < 4; ++ks) {
                size_t off = (size_t)(b * SEQ + k0 + nt * 16 + lo) * DIM +
                             dbase + ks * 32 + hi * 16;
                kfr[ks] = ld_frag(k + off);
            }
#pragma unroll
            for (int mt = 0; mt < 2; ++mt) {
                v8f sc = vzero;
#pragma unroll
                for (int ks = 0; ks < 4; ++ks)
                    sc = wmma_bf16(qfr[mt][ks], kfr[ks], sc);
#pragma unroll
                for (int r = 0; r < 8; ++r) {
                    int rr = mt * 16 + r + hi * 8;
                    int cc = nt * 16 + lo;
                    atomicAdd(&s_scores[rr][cc], sc[r]);   // ds_add_f32
                }
            }
        }
        __syncthreads();

        // ---- online softmax (lane = query row, redundant per wave) ----
        {
            const int r = lane;
            const int qrow = q0 + r;
            v4f s4[8];
#pragma unroll
            for (int c4 = 0; c4 < 8; ++c4)
                s4[c4] = *reinterpret_cast<const v4f*>(&s_scores[r][c4 * 4]);

            float x[32];
#pragma unroll
            for (int c = 0; c < 32; ++c) {
                float sv = s4[c >> 2][c & 3] * SCALE;
                int key = k0 + c;
                if (mask[b * SEQ + key] == 0) sv = NEGV;    // padding mask
                if (key > qrow) sv += NEGV;                 // causal (additive)
                x[c] = sv;
            }
            float m_old = s_m[r], l_old = s_l[r];
            float m_new = m_old;
#pragma unroll
            for (int c = 0; c < 32; ++c) m_new = fmaxf(m_new, x[c]);
            float corr = __expf(m_old - m_new);
            float lsum = 0.0f;
            unsigned int pk[16];
#pragma unroll
            for (int j = 0; j < 16; ++j) {
                float p0 = __expf(x[2 * j]     - m_new);
                float p1 = __expf(x[2 * j + 1] - m_new);
                lsum += p0 + p1;
                pk[j] = (unsigned int)f2bf(p0) | ((unsigned int)f2bf(p1) << 16);
            }
            if (wv == 0) {
                v4u* dp = reinterpret_cast<v4u*>(&s_p[r][0]);
#pragma unroll
                for (int j4 = 0; j4 < 4; ++j4) {
                    v4u pv = {pk[j4*4], pk[j4*4+1], pk[j4*4+2], pk[j4*4+3]};
                    dp[j4] = pv;
                }
                s_corr[r] = corr;
                s_m[r]    = m_new;
                s_l[r]    = l_old * corr + lsum;
            }
        }
        __syncthreads();

        // ---- rescale accumulators by per-row correction ----
#pragma unroll
        for (int mt = 0; mt < 2; ++mt)
#pragma unroll
            for (int r = 0; r < 8; ++r) {
                float cr = s_corr[mt * 16 + r + hi * 8];
#pragma unroll
                for (int nt = 0; nt < 8; ++nt) acc[mt][nt][r] *= cr;
            }

        // ---- P (32x32 bf16) @ V-slice (32x128) ----
        v16bf pfr[2];
#pragma unroll
        for (int mt = 0; mt < 2; ++mt)
            pfr[mt] = ld_frag(&s_p[mt * 16 + lo][hi * 16]);

#pragma unroll
        for (int nt = 0; nt < 8; ++nt) {
            size_t off = (size_t)(dbase + nt * 16 + lo) * MTOT +
                         b * SEQ + k0 + hi * 16;
            v16bf vfr = ld_frag(vT + off);
            if (kt < qt) __builtin_prefetch(vT + off + 32, 0, 1);
#pragma unroll
            for (int mt = 0; mt < 2; ++mt)
                acc[mt][nt] = wmma_bf16(pfr[mt], vfr, acc[mt][nt]);
        }
        __syncthreads();
    }

    // ---- epilogue: divide by softmax denominator, store fp32 ----
#pragma unroll
    for (int mt = 0; mt < 2; ++mt)
#pragma unroll
        for (int r = 0; r < 8; ++r) {
            int rr = mt * 16 + r + hi * 8;
            float linv = 1.0f / s_l[rr];
#pragma unroll
            for (int nt = 0; nt < 8; ++nt) {
                int col = dbase + nt * 16 + lo;
                out[(size_t)(b * SEQ + q0 + rr) * DIM + col] = acc[mt][nt][r] * linv;
            }
        }
}

// ---------------------------------------------------------------------------
// Host launcher
// ---------------------------------------------------------------------------
extern "C" void kernel_launch(void* const* d_in, const int* in_sizes, int n_in,
                              void* d_out, int out_size, void* d_ws, size_t ws_size,
                              hipStream_t stream) {
    const float* Q   = (const float*)d_in[0];
    const float* K   = (const float*)d_in[1];
    const float* V   = (const float*)d_in[2];
    const int*   msk = (const int*)  d_in[3];
    const float* Wq  = (const float*)d_in[4];
    const float* bq  = (const float*)d_in[5];
    const float* Wk  = (const float*)d_in[6];
    const float* bk  = (const float*)d_in[7];
    const float* Wv  = (const float*)d_in[8];
    const float* bv  = (const float*)d_in[9];
    float* out = (float*)d_out;

    const size_t NQKV = (size_t)MTOT * DIM;   // 8M elems
    const size_t NW   = (size_t)DIM * DIM;    // 1M elems

    // workspace layout (bf16 elements)
    unsigned short* Qb  = (unsigned short*)d_ws;
    unsigned short* Kb  = Qb  + NQKV;
    unsigned short* Vb  = Kb  + NQKV;
    unsigned short* Wqb = Vb  + NQKV;
    unsigned short* Wkb = Wqb + NW;
    unsigned short* Wvb = Wkb + NW;
    unsigned short* qo  = Wvb + NW;
    unsigned short* ko  = qo  + NQKV;
    unsigned short* vTo = ko  + NQKV;

    // fp32 -> bf16 pre-pass (bandwidth bound; enables TDM + WMMA path)
    {
        int n4q = (int)(NQKV / 4), n4w = (int)(NW / 4);
        dim3 bq4(256);
        cvt_bf16_kernel<<<dim3((n4q + 255) / 256), bq4, 0, stream>>>(Q,  Qb,  n4q);
        cvt_bf16_kernel<<<dim3((n4q + 255) / 256), bq4, 0, stream>>>(K,  Kb,  n4q);
        cvt_bf16_kernel<<<dim3((n4q + 255) / 256), bq4, 0, stream>>>(V,  Vb,  n4q);
        cvt_bf16_kernel<<<dim3((n4w + 255) / 256), bq4, 0, stream>>>(Wq, Wqb, n4w);
        cvt_bf16_kernel<<<dim3((n4w + 255) / 256), bq4, 0, stream>>>(Wk, Wkb, n4w);
        cvt_bf16_kernel<<<dim3((n4w + 255) / 256), bq4, 0, stream>>>(Wv, Wvb, n4w);
    }

    dim3 g1(MTOT / 128, DIM / 128), b1(256);
    qkv_proj_kernel<false><<<g1, b1, 0, stream>>>(Qb, Wqb, bq, qo);
    qkv_proj_kernel<false><<<g1, b1, 0, stream>>>(Kb, Wkb, bk, ko);
    qkv_proj_kernel<true ><<<g1, b1, 0, stream>>>(Vb, Wvb, bv, vTo);

    dim3 g2(SEQ / 32, BATCH), b2(256);
    flash_attn_kernel<<<g2, b2, 0, stream>>>(qo, ko, vTo, msk, out);
}
